// MyMerge_67216238182733
// MI455X (gfx1250) — matrix operand order repaired and checked by reading, not compiled, not run on hardware
//
#include <hip/hip_runtime.h>
#include <hip/hip_bf16.h>
#include <stdint.h>

// Problem constants (match reference)
#define BB 32
#define LL 512
#define DD 512
#define SS 8

typedef __attribute__((ext_vector_type(16))) __bf16 v16bf;
typedef __attribute__((ext_vector_type(8)))  __bf16 v8bf;
typedef __attribute__((ext_vector_type(8)))  float  v8f;
typedef __attribute__((ext_vector_type(4)))  unsigned int u32x4;
typedef __attribute__((ext_vector_type(8)))  int i32x8;
typedef __attribute__((ext_vector_type(4)))  int i32x4;

__device__ __forceinline__ unsigned short f32_to_bf16_bits(float f) {
    union { float f; unsigned int u; } cv; cv.f = f;
    unsigned int u = cv.u;
    u += 0x7fffu + ((u >> 16) & 1u);   // round-to-nearest-even
    return (unsigned short)(u >> 16);
}

__device__ __forceinline__ float sigmoidf(float x) {
    return 1.0f / (1.0f + __expf(-x));
}

// Load a 16-element bf16 fragment from two 8-element (16B) chunks.
__device__ __forceinline__ v16bf load_frag2(const unsigned short* p_lo,
                                            const unsigned short* p_hi) {
    v8bf lo = *reinterpret_cast<const v8bf*>(p_lo);
    v8bf hi = *reinterpret_cast<const v8bf*>(p_hi);
    return __builtin_shufflevector(lo, hi, 0,1,2,3,4,5,6,7,8,9,10,11,12,13,14,15);
}

// ---------------------------------------------------------------------------
// TENSORcnt waits (SOPP s_wait_tensorcnt)
// ---------------------------------------------------------------------------
__device__ __forceinline__ void wait_tensorcnt(int n_imm_0_or_1) {
#if __has_builtin(__builtin_amdgcn_s_wait_tensorcnt)
    if (n_imm_0_or_1 == 0) __builtin_amdgcn_s_wait_tensorcnt(0);
    else                   __builtin_amdgcn_s_wait_tensorcnt(1);
#else
    if (n_imm_0_or_1 == 0) asm volatile("s_wait_tensorcnt 0x0" ::: "memory");
    else                   asm volatile("s_wait_tensorcnt 0x1" ::: "memory");
#endif
}

// ---------------------------------------------------------------------------
// Generic TDM tile load (bf16 / 2-byte elements), 2-D (td2==0) or 3-D tile.
//   dim0: contiguous, tile td0 elements, tensor extent tensor_d0
//   dim1: stride s0 elements between lines,  tile td1, extent tensor_d1
//   dim2: stride s1 elements between planes, tile td2, extent tensor_d2
// gptr = global tile start; sptr = LDS destination (tile packed linearly).
// Descriptor per CDNA5 ISA sec.8 (D# groups 0..3).
// ---------------------------------------------------------------------------
__device__ __forceinline__ void tdm_load_tile(const unsigned short* gptr,
                                              unsigned short* sptr,
                                              unsigned td0, unsigned td1, unsigned td2,
                                              unsigned tensor_d0, unsigned tensor_d1,
                                              unsigned tensor_d2,
                                              unsigned long long s0,
                                              unsigned long long s1) {
#if __has_builtin(__builtin_amdgcn_tensor_load_to_lds)
    unsigned long long gaddr = (unsigned long long)(uintptr_t)gptr;
    unsigned int lds_addr    = (unsigned int)(uintptr_t)sptr;   // LDS byte offset
    u32x4 g0;
    g0[0] = 1u;                                       // count=1 (valid), user mode
    g0[1] = lds_addr;                                 // lds_addr
    g0[2] = (unsigned int)(gaddr & 0xffffffffull);    // global_addr[31:0]
    g0[3] = (unsigned int)((gaddr >> 32) & 0x01ffffffull) | (2u << 30); // [56:32], type=2
    i32x8 g1;
    g1[0] = 0x10000;                                  // mask=0, data_size=1 (2 bytes)
    g1[1] = (int)((tensor_d0 & 0xffffu) << 16);                       // dim0[15:0]
    g1[2] = (int)(((tensor_d0 >> 16) & 0xffffu) |
                  ((tensor_d1 & 0xffffu) << 16));                     // dim0 hi | dim1 lo
    g1[3] = (int)(((tensor_d1 >> 16) & 0xffffu) | (td0 << 16));       // dim1 hi | tile0
    g1[4] = (int)(td1 | (td2 << 16));                                 // tile1 | tile2
    g1[5] = (int)(s0 & 0xffffffffull);                                // stride0 lo
    g1[6] = (int)(((s0 >> 32) & 0xffffull) | ((s1 & 0xffffull) << 16));
    g1[7] = (int)((s1 >> 16) & 0xffffffffull);                        // stride1 hi
    i32x4 g2;
    g2[0] = (int)tensor_d2;                           // tensor_dim2 extent
    g2[1] = 0;                                        // tensor_dim3 unused
    g2[2] = 0;                                        // tensor_dim2_stride (dim3 unused)
    g2[3] = 0;                                        // tile_dim3 = 0
    i32x4 g3; g3[0] = 0; g3[1] = 0; g3[2] = 0; g3[3] = 0;
#if __clang_major__ >= 23
    i32x8 gz; gz[0]=0; gz[1]=0; gz[2]=0; gz[3]=0; gz[4]=0; gz[5]=0; gz[6]=0; gz[7]=0;
    __builtin_amdgcn_tensor_load_to_lds(g0, g1, g2, g3, gz, 0);
#else
    __builtin_amdgcn_tensor_load_to_lds(g0, g1, g2, g3, 0);
#endif
#else
    // Fallback: one wave copies the tile with plain loads/stores (16B chunks).
    int lane = threadIdx.x & 31;
    unsigned nrows = td1 * (td2 ? td2 : 1u);
    unsigned cpr   = td0 >> 3;                        // 8-element chunks per row
    for (unsigned c = lane; c < nrows * cpr; c += 32) {
        unsigned r  = c / cpr;
        unsigned e8 = (c % cpr) * 8;
        unsigned r1 = r % td1;
        unsigned r2 = r / td1;
        *reinterpret_cast<v8bf*>(sptr + (size_t)r * td0 + e8) =
            *reinterpret_cast<const v8bf*>(gptr + (size_t)r1 * s0 +
                                           (size_t)r2 * s1 + e8);
    }
#endif
}

// ---------------------------------------------------------------------------
// Kernel: f32 -> bf16 elementwise convert
// ---------------------------------------------------------------------------
__global__ void cvt_bf16_kernel(const float* __restrict__ src,
                                unsigned short* __restrict__ dst, int n) {
    int i = blockIdx.x * blockDim.x + threadIdx.x;
    if (i < n) dst[i] = f32_to_bf16_bits(src[i]);
}

// ---------------------------------------------------------------------------
// Kernel: Mr[k][d][e] (f32) -> MrT[k][e][d] (bf16) so stage-1 WMMA B-fragments
// (K = d, contiguous) are linear loads / TDM rows.
// ---------------------------------------------------------------------------
__global__ void cvt_mr_transpose_kernel(const float* __restrict__ Mr,
                                        unsigned short* __restrict__ MrT) {
    int idx = blockIdx.x * blockDim.x + threadIdx.x;
    if (idx >= SS * DD * DD) return;
    int k   = idx / (DD * DD);
    int rem = idx - k * DD * DD;
    int d   = rem / DD;
    int e   = rem - d * DD;
    MrT[((size_t)k * DD + e) * DD + d] = f32_to_bf16_bits(Mr[idx]);
}

// ---------------------------------------------------------------------------
// Kernel: small projections P1 = arg1@Wg[:D], P2 = arg2@Wg[D:],
//                           Q1 = arg1@V[:D],  Q2 = arg2@V[D:]   ([B,L,S] f32)
// ---------------------------------------------------------------------------
__global__ void proj_kernel(const float* __restrict__ arg1,
                            const float* __restrict__ arg2,
                            const float* __restrict__ Wg,
                            const float* __restrict__ V,
                            float* __restrict__ P1, float* __restrict__ Q1,
                            float* __restrict__ P2, float* __restrict__ Q2) {
    int tid = blockIdx.x * blockDim.x + threadIdx.x;
    if (tid >= BB * LL * SS) return;
    int s   = tid & (SS - 1);
    int row = tid >> 3;                 // b*L + i
    const float* a1 = arg1 + (size_t)row * DD;
    const float* a2 = arg2 + (size_t)row * DD;
    float p1 = 0.f, q1 = 0.f, p2 = 0.f, q2 = 0.f;
    for (int d = 0; d < DD; ++d) {
        float x1 = a1[d], x2 = a2[d];
        p1 += x1 * Wg[d * SS + s];
        p2 += x2 * Wg[(DD + d) * SS + s];
        q1 += x1 * V[d * SS + s];
        q2 += x2 * V[(DD + d) * SS + s];
    }
    P1[tid] = p1; Q1[tid] = q1; P2[tid] = p2; Q2[tid] = q2;
}

// ---------------------------------------------------------------------------
// Stage 1 (TDM + LDS): A1[b,k] = arg1[b] (LxD) @ Mr[k] (DxD).
// 8 waves/block cover 8 i-tiles sharing the same (b,k,e-supertile); the MrT
// chunk [e0..e0+63][d0..d0+31] (4 KB) is TDM-staged into double-buffered LDS.
// Each wave: 1 global A-frag + 4 LDS B-frags -> 4 WMMAs per K-chunk.
// Output layout: A1[((b*S + k)*L + i)*D + e].
// ---------------------------------------------------------------------------
__global__ void __launch_bounds__(256)
stage1_gemm_kernel(const unsigned short* __restrict__ arg1_bf,
                   const unsigned short* __restrict__ MrT_bf,
                   unsigned short* __restrict__ A1_bf) {
    extern __shared__ unsigned short smem[];   // 2 x 2048 bf16 = 8 KB
    const int tid  = threadIdx.x;
    const int wave = tid >> 5;
    const int lane = tid & 31;
    const int row  = lane & 15;
    const int hs   = lane >> 4;
    const int e0   = blockIdx.x * 64;
    const int i0   = blockIdx.y * 128 + wave * 16;
    const int bk   = blockIdx.z;
    const int b    = bk / SS;
    const int k    = bk % SS;

    const unsigned short* Arow  = arg1_bf + (size_t)(b * LL + i0 + row) * DD;
    const unsigned short* Bbase = MrT_bf + ((size_t)k * DD + e0) * DD;  // rows = e

    if (wave == 0)
        tdm_load_tile(Bbase, smem, 32, 64, 0, DD, DD, 0, DD, 0);

    v8f acc[4] = {};
    for (int n = 0; n < DD / 32; ++n) {
        if (wave == 0) {
            if (n + 1 < DD / 32) {
                tdm_load_tile(Bbase + (n + 1) * 32, smem + ((n + 1) & 1) * 2048,
                              32, 64, 0, DD, DD, 0, DD, 0);
                wait_tensorcnt(1);
            } else {
                wait_tensorcnt(0);
            }
        }
        __syncthreads();

        const int d0 = n * 32;
        v16bf afrag = load_frag2(Arow + d0 + hs * 8, Arow + d0 + 16 + hs * 8);
        const unsigned short* sbuf = smem + (n & 1) * 2048;   // [64 e][32 d]
#pragma unroll
        for (int t = 0; t < 4; ++t) {
            const unsigned short* br = sbuf + (t * 16 + row) * 32;
            v16bf bfrag = load_frag2(br + hs * 8, br + 16 + hs * 8);
            acc[t] = __builtin_amdgcn_wmma_f32_16x16x32_bf16(
                false, afrag, false, bfrag, (short)0, acc[t], false, false);
        }
        __syncthreads();
    }

    unsigned short* outb = A1_bf + (size_t)((b * SS + k) * LL) * DD;
#pragma unroll
    for (int t = 0; t < 4; ++t) {
#pragma unroll
        for (int r = 0; r < 8; ++r) {
            int i = i0 + hs * 8 + r;
            int e = e0 + t * 16 + row;
            outb[(size_t)i * DD + e] = f32_to_bf16_bits(acc[t][r]);
        }
    }
}

// ---------------------------------------------------------------------------
// Stage 2 (fused, TDM + LDS): 8 waves/block share one 16-row i-tile of A1
// (all 8 k-slices) TDM-staged into double-buffered LDS. Each wave owns one
// 16x16 j-tile; per K-chunk: 1 global B-frag + 8 LDS A-frags -> 8 WMMAs.
// Epilogue applies both sigmoid gates, U-dot and final sigmoid.
// ---------------------------------------------------------------------------
__global__ void __launch_bounds__(256)
stage2_fused_kernel(const unsigned short* __restrict__ A1_bf,
                    const unsigned short* __restrict__ arg2_bf,
                    const float* __restrict__ P1, const float* __restrict__ Q1,
                    const float* __restrict__ P2, const float* __restrict__ Q2,
                    const float* __restrict__ Bg, const float* __restrict__ bconst,
                    const float* __restrict__ U, float* __restrict__ out) {
    extern __shared__ unsigned short smem[];   // 2 x 4096 bf16 = 16 KB
    const int tid  = threadIdx.x;
    const int wave = tid >> 5;
    const int lane = tid & 31;
    const int col  = lane & 15;
    const int hs   = lane >> 4;
    const int j0   = blockIdx.x * 128 + wave * 16;
    const int i0   = blockIdx.y * 16;
    const int b    = blockIdx.z;

    const unsigned short* Brow  = arg2_bf + (size_t)(b * LL + j0 + col) * DD;
    const unsigned short* Abase =
        A1_bf + ((size_t)(b * SS) * LL + i0) * DD;   // A1[b][0][i0][0]

    if (wave == 0)
        tdm_load_tile(Abase, smem, 32, 16, 8, DD, LL, SS,
                      DD, (unsigned long long)LL * DD);

    v8f acc[SS] = {};
    for (int n = 0; n < DD / 32; ++n) {
        if (wave == 0) {
            if (n + 1 < DD / 32) {
                tdm_load_tile(Abase + (size_t)(n + 1) * 32,
                              smem + ((n + 1) & 1) * 4096,
                              32, 16, 8, DD, LL, SS,
                              DD, (unsigned long long)LL * DD);
                wait_tensorcnt(1);     // current chunk done, prefetch in flight
            } else {
                wait_tensorcnt(0);
            }
        }
        __syncthreads();               // LDS buffer (n&1) valid for all waves

        const int e0 = n * 32;
        v16bf bfrag = load_frag2(Brow + e0 + hs * 8, Brow + e0 + 16 + hs * 8);
        const unsigned short* sbuf = smem + (n & 1) * 4096;   // [k][row][32 e]
#pragma unroll
        for (int k = 0; k < SS; ++k) {
            const unsigned short* ar = sbuf + (k * 16 + col) * 32;
            v16bf afrag = load_frag2(ar + hs * 8, ar + 16 + hs * 8);
            acc[k] = __builtin_amdgcn_wmma_f32_16x16x32_bf16(
                false, afrag, false, bfrag, (short)0, acc[k], false, false);
        }
        __syncthreads();               // all reads done before buffer reuse
    }

    // Epilogue: gated blend of bilinear + sigmoid-interaction terms.
    float bg_l[SS], bc_l[SS], u_l[SS], p2_l[SS], q2_l[SS];
    const int j = j0 + col;
#pragma unroll
    for (int s = 0; s < SS; ++s) {
        bg_l[s] = Bg[s];
        bc_l[s] = bconst[s];
        u_l[s]  = U[s];
        p2_l[s] = P2[(size_t)(b * LL + j) * SS + s];
        q2_l[s] = Q2[(size_t)(b * LL + j) * SS + s];
    }
#pragma unroll
    for (int r = 0; r < 8; ++r) {
        const int i = i0 + hs * 8 + r;
        const float* p1 = P1 + (size_t)(b * LL + i) * SS;
        const float* q1 = Q1 + (size_t)(b * LL + i) * SS;
        float sum = 0.f;
#pragma unroll
        for (int s = 0; s < SS; ++s) {
            float g  = sigmoidf(p1[s] + p2_l[s] + bg_l[s]);
            float sv = sigmoidf(q1[s] + q2_l[s]);
            sum += (acc[s][r] * g + sv * (1.0f - g) + bc_l[s]) * u_l[s];
        }
        out[((size_t)b * LL + i) * LL + j] = sigmoidf(sum);
    }
}

// ---------------------------------------------------------------------------
extern "C" void kernel_launch(void* const* d_in, const int* in_sizes, int n_in,
                              void* d_out, int out_size, void* d_ws, size_t ws_size,
                              hipStream_t stream) {
    (void)in_sizes; (void)n_in; (void)out_size; (void)ws_size;
    const float* arg1 = (const float*)d_in[0];
    const float* arg2 = (const float*)d_in[1];
    const float* Wg   = (const float*)d_in[2];
    const float* Bg   = (const float*)d_in[3];
    const float* Mr   = (const float*)d_in[4];
    const float* V    = (const float*)d_in[5];
    const float* bcst = (const float*)d_in[6];
    const float* U    = (const float*)d_in[7];
    float* out = (float*)d_out;

    char* ws = (char*)d_ws;
    size_t off = 0;
    auto carve = [&](size_t bytes) -> char* {
        char* p = ws + off;
        off += (bytes + 255) & ~(size_t)255;
        return p;
    };
    unsigned short* arg1_bf = (unsigned short*)carve((size_t)BB * LL * DD * 2);
    unsigned short* arg2_bf = (unsigned short*)carve((size_t)BB * LL * DD * 2);
    unsigned short* MrT_bf  = (unsigned short*)carve((size_t)SS * DD * DD * 2);
    unsigned short* A1_bf   = (unsigned short*)carve((size_t)BB * SS * LL * DD * 2);
    float* P1 = (float*)carve((size_t)BB * LL * SS * 4);
    float* Q1 = (float*)carve((size_t)BB * LL * SS * 4);
    float* P2 = (float*)carve((size_t)BB * LL * SS * 4);
    float* Q2 = (float*)carve((size_t)BB * LL * SS * 4);

    const int n_arg = BB * LL * DD;
    cvt_bf16_kernel<<<(n_arg + 255) / 256, 256, 0, stream>>>(arg1, arg1_bf, n_arg);
    cvt_bf16_kernel<<<(n_arg + 255) / 256, 256, 0, stream>>>(arg2, arg2_bf, n_arg);

    const int n_mr = SS * DD * DD;
    cvt_mr_transpose_kernel<<<(n_mr + 255) / 256, 256, 0, stream>>>(Mr, MrT_bf);

    const int n_proj = BB * LL * SS;
    proj_kernel<<<(n_proj + 255) / 256, 256, 0, stream>>>(arg1, arg2, Wg, V,
                                                          P1, Q1, P2, Q2);

    // Stage 1: 8 waves/block over 8 i-tiles; 8 KB dynamic LDS (TDM B tiles).
    stage1_gemm_kernel<<<dim3(DD / 64, LL / 128, BB * SS), 256, 8192, stream>>>(
        arg1_bf, MrT_bf, A1_bf);

    // Stage 2: 8 waves/block over 8 j-tiles; 16 KB dynamic LDS (TDM A tiles).
    stage2_fused_kernel<<<dim3(LL / 128, LL / 16, BB), 256, 16384, stream>>>(
        A1_bf, arg2_bf, P1, Q1, P2, Q2, Bg, bcst, U, out);
}
